// GConv_3221225472297
// MI455X (gfx1250) — compile-verified, matrix operand order checked by reading, not run on hardware
//
#include <hip/hip_runtime.h>
#include <hip/hip_bf16.h>
#include <stdint.h>

// ---------------------------------------------------------------------------
// GIN (3 layers): scatter-add aggregation + WMMA f16 MLP (f32 accumulate) +
// fused BatchNorm statistics + BN apply + segment-sum pooling.
// Target: gfx1250 (CDNA5), wave32, V_WMMA_F32_16X16X32_F16.
// Persistent-grid MLP: weights packed to LDS once per block, B fragments
// hoisted to registers once per wave, grid-stride over 16-row tiles.
// ---------------------------------------------------------------------------

typedef __attribute__((ext_vector_type(16))) _Float16 v16h;
typedef __attribute__((ext_vector_type(8)))  _Float16 v8h;
typedef __attribute__((ext_vector_type(8)))  float    v8f;

#define GIN_D 64
#define NUM_GRAPHS 512
#define BN_EPS 1e-5f
#define MLP_BLOCKS 256

// ---------------------------------------------------------------- utilities
__global__ void zero_f32_kernel(float* __restrict__ p, int n) {
  int i = blockIdx.x * blockDim.x + threadIdx.x;
  if (i < n) p[i] = 0.0f;
}

__global__ void copy_f32x4_kernel(float* __restrict__ dst,
                                  const float* __restrict__ src, int n) {
  int i = (blockIdx.x * blockDim.x + threadIdx.x) * 4;
  if (i + 3 < n) {
    *(float4*)(dst + i) = *(const float4*)(src + i);
  } else {
    for (; i < n; ++i) dst[i] = src[i];
  }
}

// ------------------------------------------------------------- edge scatter
// h[dst] += z[src]; 16 lanes per edge, float4 per lane.
// Feature matrix (25.6 MB) is L2-resident, so gathers + atomics land in GL2.
__global__ void gin_scatter_kernel(const float* __restrict__ z,
                                   const long long* __restrict__ ei,
                                   float* __restrict__ h, long long E) {
  long long t = (long long)blockIdx.x * blockDim.x + threadIdx.x;
  long long e = t >> 4;
  int part = (int)(t & 15);
  if (e >= E) return;
  long long s = ei[e];       // edge_index[0][e]
  long long d = ei[E + e];   // edge_index[1][e]
  const float4 v = *(const float4*)(z + s * GIN_D + part * 4);
  float* dp = h + d * GIN_D + part * 4;
  atomicAdd(dp + 0, v.x);
  atomicAdd(dp + 1, v.y);
  atomicAdd(dp + 2, v.z);
  atomicAdd(dp + 3, v.w);
}

// ------------------------------------------------- WMMA MLP + BN statistics
// u = relu(relu(h@W1 + b1)@W2 + b2), plus per-column sum / sum-of-squares.
// One wave = one 16-row tile per loop trip. K = 64 => 2 WMMAs per 16x16
// output tile, 4 column tiles, 2 GEMMs = 16 v_wmma per tile.
__global__ __launch_bounds__(256) void gin_mlp_wmma_kernel(
    const float* h,                       // [N,64] (may alias u)
    const float* __restrict__ W1,         // [64,64] row-major (k-major)
    const float* __restrict__ b1,         // [64]
    const float* __restrict__ W2,         // [64,64]
    const float* __restrict__ b2,         // [64]
    float* u,                             // [N,64] (may alias h)
    float* __restrict__ bn_sum,           // [64]
    float* __restrict__ bn_sq,            // [64]
    int N, int num_tiles) {
  // Weights packed in exact B-fragment layout: frag f = jt*2 + kb,
  // lane slot holds the 16 halves that lane's v16h B operand needs.
  __shared__ __align__(32) _Float16 sW[2][8][32][16];   // 16 KB
  __shared__ __align__(32) _Float16 sT[8][16 * GIN_D];  // 16 KB (per-wave pads)
  __shared__ float sB1[GIN_D], sB2[GIN_D];
  __shared__ float sSum[GIN_D], sSq[GIN_D];

  // ---- pack phase: 512 (matrix, frag, lane) slots over 256 threads
  for (int s = threadIdx.x; s < 512; s += 256) {
    int m  = s >> 8;         // 0 = W1, 1 = W2
    int f  = (s >> 5) & 7;   // fragment id: jt*2 + kb
    int ln = s & 31;         // lane slot
    int jt = f >> 1, kb = f & 1;
    const float* W = m ? W2 : W1;
    int j    = jt * 16 + (ln & 15);
    int krow = (ln >> 4) * 8;
#pragma unroll
    for (int p = 0; p < 16; ++p) {
      int k = kb * 32 + (p >> 3) * 16 + krow + (p & 7);
      sW[m][f][ln][p] = (_Float16)W[k * GIN_D + j];
    }
  }
  if (threadIdx.x < GIN_D) {
    sB1[threadIdx.x]  = b1[threadIdx.x];
    sB2[threadIdx.x]  = b2[threadIdx.x];
    sSum[threadIdx.x] = 0.0f;
    sSq[threadIdx.x]  = 0.0f;
  }
  __syncthreads();

  const int wave = threadIdx.x >> 5;
  const int lane = threadIdx.x & 31;
  const int kb8   = (lane >> 4) * 8;
  const int mBase = (lane >> 4) * 8;
  const int lcol  = lane & 15;

  // ---- hoist all 16 B fragments (W1 + W2) into registers: loop-invariant
  v16h Bf1[8], Bf2[8];
#pragma unroll
  for (int f = 0; f < 8; ++f) {
    Bf1[f] = *(const v16h*)(&sW[0][f][lane][0]);
    Bf2[f] = *(const v16h*)(&sW[1][f][lane][0]);
  }
  const float bias1[4] = {sB1[0 + lcol], sB1[16 + lcol], sB1[32 + lcol], sB1[48 + lcol]};
  const float bias2[4] = {sB2[0 + lcol], sB2[16 + lcol], sB2[32 + lcol], sB2[48 + lcol]};

  const int tstride = gridDim.x * 8;
  _Float16* tp = &sT[wave][0];

  for (int tile = blockIdx.x * 8 + wave; tile < num_tiles; tile += tstride) {
    const int row0 = tile * 16;
    const int r    = row0 + lcol;
    const bool valid = (r < N);
    const float* hp = h + (size_t)r * GIN_D;

    // Prefetch next tile's activation row into cache (global_prefetch_b8).
    if (tile + tstride < num_tiles)
      __builtin_prefetch(h + (size_t)(row0 + tstride * 16 + lcol) * GIN_D, 0, 1);

    // ---- A fragments for GEMM1 from global h (f32 -> f16)
    v16h a1[2];
#pragma unroll
    for (int kb = 0; kb < 2; ++kb) {
      float tmp[16];
      if (valid) {
        const int base = kb * 32 + kb8;
        const float4 v0 = *(const float4*)(hp + base + 0);
        const float4 v1 = *(const float4*)(hp + base + 4);
        const float4 v2 = *(const float4*)(hp + base + 16);
        const float4 v3 = *(const float4*)(hp + base + 20);
        tmp[0]=v0.x;  tmp[1]=v0.y;  tmp[2]=v0.z;  tmp[3]=v0.w;
        tmp[4]=v1.x;  tmp[5]=v1.y;  tmp[6]=v1.z;  tmp[7]=v1.w;
        tmp[8]=v2.x;  tmp[9]=v2.y;  tmp[10]=v2.z; tmp[11]=v2.w;
        tmp[12]=v3.x; tmp[13]=v3.y; tmp[14]=v3.z; tmp[15]=v3.w;
      } else {
#pragma unroll
        for (int p = 0; p < 16; ++p) tmp[p] = 0.0f;
      }
#pragma unroll
      for (int p = 0; p < 16; ++p) a1[kb][p] = (_Float16)tmp[p];
    }

    // ---- GEMM1: t = relu(h @ W1 + b1) -> stage f16 to per-wave LDS pad
#pragma unroll
    for (int jt = 0; jt < 4; ++jt) {
      v8f c = {};
      c = __builtin_amdgcn_wmma_f32_16x16x32_f16(false, a1[0], false, Bf1[jt * 2 + 0],
                                                 (short)0, c, false, false);
      c = __builtin_amdgcn_wmma_f32_16x16x32_f16(false, a1[1], false, Bf1[jt * 2 + 1],
                                                 (short)0, c, false, false);
      const int col = jt * 16 + lcol;
#pragma unroll
      for (int rr = 0; rr < 8; ++rr) {
        float t = c[rr] + bias1[jt];
        t = t > 0.0f ? t : 0.0f;
        tp[(mBase + rr) * GIN_D + col] = (_Float16)t;  // C-layout -> row-major
      }
    }

    // ---- A fragments for GEMM2 from LDS pad (same-wave LDS is in-order)
    v16h a2[2];
#pragma unroll
    for (int kb = 0; kb < 2; ++kb) {
      const int base = lcol * GIN_D + kb * 32 + kb8;
      const v8h lo = *(const v8h*)(&sT[wave][base]);
      const v8h hi = *(const v8h*)(&sT[wave][base + 16]);
#pragma unroll
      for (int p = 0; p < 8; ++p) { a2[kb][p] = lo[p]; a2[kb][8 + p] = hi[p]; }
    }

    // ---- GEMM2: u = relu(t @ W2 + b2); store + BN partial sums
#pragma unroll
    for (int jt = 0; jt < 4; ++jt) {
      v8f c = {};
      c = __builtin_amdgcn_wmma_f32_16x16x32_f16(false, a2[0], false, Bf2[jt * 2 + 0],
                                                 (short)0, c, false, false);
      c = __builtin_amdgcn_wmma_f32_16x16x32_f16(false, a2[1], false, Bf2[jt * 2 + 1],
                                                 (short)0, c, false, false);
      const int col = jt * 16 + lcol;
      float s = 0.0f, q = 0.0f;
#pragma unroll
      for (int rr = 0; rr < 8; ++rr) {
        const int grow = row0 + mBase + rr;
        float t = c[rr] + bias2[jt];
        t = t > 0.0f ? t : 0.0f;
        if (grow < N) {
          u[(size_t)grow * GIN_D + col] = t;
          s += t;
          q += t * t;
        }
      }
      atomicAdd(&sSum[col], s);  // ds_add_f32
      atomicAdd(&sSq[col],  q);
    }
  }

  __syncthreads();
  if (threadIdx.x < GIN_D) {
    atomicAdd(&bn_sum[threadIdx.x], sSum[threadIdx.x]);
    atomicAdd(&bn_sq[threadIdx.x],  sSq[threadIdx.x]);
  }
}

// ----------------------------------------------- BN coefficients (64 threads)
// a = gamma * rsqrt(var + eps), c = beta - mean * a; also resets stats.
__global__ void bn_coef_kernel(float* __restrict__ bn_sum,
                               float* __restrict__ bn_sq,
                               const float* __restrict__ gamma,
                               const float* __restrict__ beta,
                               float* __restrict__ cA,
                               float* __restrict__ cC, int N) {
  int d = threadIdx.x;
  if (d < GIN_D) {
    const float inv_n = 1.0f / (float)N;
    const float mean = bn_sum[d] * inv_n;
    const float var  = bn_sq[d] * inv_n - mean * mean;
    const float a    = gamma[d] * rsqrtf(var + BN_EPS);
    cA[d] = a;
    cC[d] = beta[d] - mean * a;
    bn_sum[d] = 0.0f;  // ready for next layer
    bn_sq[d]  = 0.0f;
  }
}

// -------------------------------------- BN apply + pooling (+ seed next h)
__global__ void bn_pool_kernel(const float* u,           // may alias hnext
                               const float* __restrict__ cA,
                               const float* __restrict__ cC,
                               const long long* __restrict__ batch,
                               float* __restrict__ zout,
                               float* hnext,              // nullptr on last layer
                               float* __restrict__ g,     // [512, 3*64]
                               int layer, int N) {
  int i = blockIdx.x * blockDim.x + threadIdx.x;
  if (i >= N * GIN_D) return;
  const int node = i >> 6;
  const int d    = i & 63;
  const float zv = u[i] * cA[d] + cC[d];
  zout[i] = zv;
  if (hnext) hnext[i] = zv;  // next layer's h starts at z (GIN eps=0)
  const long long gi = batch[node];
  atomicAdd(&g[gi * (3 * GIN_D) + layer * GIN_D + d], zv);
}

// --------------------------------------------------------------------- host
extern "C" void kernel_launch(void* const* d_in, const int* in_sizes, int n_in,
                              void* d_out, int out_size, void* d_ws, size_t ws_size,
                              hipStream_t stream) {
  const float*      x     = (const float*)d_in[0];
  const long long*  ei    = (const long long*)d_in[1];  // int64 [2,E]
  const long long*  batch = (const long long*)d_in[2];  // int64 [N]
  const float*      W1    = (const float*)d_in[3];      // [3,64,64]
  const float*      b1    = (const float*)d_in[4];      // [3,64]
  const float*      W2    = (const float*)d_in[5];
  const float*      b2    = (const float*)d_in[6];
  const float*      gamma = (const float*)d_in[7];
  const float*      beta  = (const float*)d_in[8];

  const int       N = in_sizes[0] / GIN_D;
  const long long E = (long long)in_sizes[1] / 2;
  float* g = (float*)d_out;

  // Workspace layout: h (aliased as u) | z | bn stats + coefs.
  float* h_buf  = (float*)d_ws;
  float* z_buf  = h_buf + (size_t)N * GIN_D;
  float* bn_sum = z_buf + (size_t)N * GIN_D;
  float* bn_sq  = bn_sum + GIN_D;
  float* cA     = bn_sq + GIN_D;
  float* cC     = cA + GIN_D;

  const int nelem = N * GIN_D;
  const int tiles = (N + 15) / 16;
  const int groups = (tiles + 7) / 8;

  zero_f32_kernel<<<(out_size + 255) / 256, 256, 0, stream>>>(g, out_size);
  zero_f32_kernel<<<1, 128, 0, stream>>>(bn_sum, 2 * GIN_D);
  copy_f32x4_kernel<<<(nelem / 4 + 255) / 256, 256, 0, stream>>>(h_buf, x, nelem);

  const long long sthreads = E * 16;
  const unsigned  sgrid = (unsigned)((sthreads + 255) / 256);
  const dim3 mlp_grid((unsigned)(groups < MLP_BLOCKS ? groups : MLP_BLOCKS));

  for (int l = 0; l < 3; ++l) {
    const float* zin = (l == 0) ? x : z_buf;
    gin_scatter_kernel<<<sgrid, 256, 0, stream>>>(zin, ei, h_buf, E);
    gin_mlp_wmma_kernel<<<mlp_grid, 256, 0, stream>>>(
        h_buf, W1 + l * GIN_D * GIN_D, b1 + l * GIN_D,
        W2 + l * GIN_D * GIN_D, b2 + l * GIN_D,
        h_buf /* u aliases h: tile reads precede tile writes */,
        bn_sum, bn_sq, N, tiles);
    bn_coef_kernel<<<1, 64, 0, stream>>>(bn_sum, bn_sq, gamma + l * GIN_D,
                                         beta + l * GIN_D, cA, cC, N);
    bn_pool_kernel<<<(nelem + 255) / 256, 256, 0, stream>>>(
        h_buf, cA, cC, batch, z_buf,
        (l < 2) ? h_buf : (float*)nullptr, g, l, N);
  }
}
